// HypLoRALayer_81870666596449
// MI455X (gfx1250) — matrix operand order repaired and checked by reading, not compiled, not run on hardware
//
#include <hip/hip_runtime.h>
#include <hip/hip_bf16.h>
#include <math.h>

typedef __attribute__((ext_vector_type(16))) __bf16 v16bf;
typedef __attribute__((ext_vector_type(8)))  __bf16 v8bf;
typedef __attribute__((ext_vector_type(8)))  float  v8f;

#define D_DIM 4096
#define M_DIM 8192
#define R_DIM 8
#define BM 128
#define BN 128
#define BK 32
#define KPAD 40   // 80B row stride -> conflict-free 16B fragment gathers

// scalar factor s(n) such that log0(exp0(v)) == s(|v|) * v  (faithful to reference clamps)
__device__ inline float hyp_scale(float n, float sqrtK) {
    const float EPS = 1e-7f;
    float nc    = fminf(n, 1.5f);
    float ratio = nc / fmaxf(n, EPS);
    float t     = sqrtK * nc;
    float coeff = sinhf(t) / (sqrtK * fmaxf(nc, EPS));
    float d     = acoshf(fmaxf(coshf(t), 1.0f + EPS)) / sqrtK;
    float ns    = coeff * ratio * n;
    return d * coeff * ratio / fmaxf(ns, EPS);
}

// ---------------- kernel 0: K scalars + Gram matrix G = B @ B^T (8x8) ----------------
__global__ __launch_bounds__(256) void prep_kernel(const float* __restrict__ B2,
                                                   const float* __restrict__ logK,
                                                   float* __restrict__ G,
                                                   float* __restrict__ kparams) {
    __shared__ float gr[64][4];
    int tid = threadIdx.x;
    int idx = tid & 63, chunk = tid >> 6;
    int r1 = idx >> 3, r2 = idx & 7;
    float g = 0.f;
    const float* p1 = B2 + r1 * D_DIM + chunk * 1024;
    const float* p2 = B2 + r2 * D_DIM + chunk * 1024;
    for (int o = 0; o < 1024; ++o) g += p1[o] * p2[o];
    gr[idx][chunk] = g;
    __syncthreads();
    if (tid < 64) G[tid] = gr[tid][0] + gr[tid][1] + gr[tid][2] + gr[tid][3];
    if (tid == 0) {
        float K = fminf(fmaxf(expf(logK[0]), 1e-4f), 10.0f);
        kparams[0] = K;
        kparams[1] = sqrtf(K);
    }
}

// ---------------- kernel 1: per-row norm, t = s1*(x@A), srow = SCALE*s2 ----------------
__global__ __launch_bounds__(256) void row_kernel(const float* __restrict__ x,
                                                  const float* __restrict__ A,
                                                  const float* __restrict__ G,
                                                  const float* __restrict__ kparams,
                                                  float* __restrict__ tbuf,
                                                  float* __restrict__ srow) {
    __shared__ float red[256][9];
    __shared__ float fin[9];
    int row = blockIdx.x;
    int tid = threadIdx.x;
    const float* xr = x + (size_t)row * D_DIM;
    float ss = 0.f;
    float u[8] = {0.f,0.f,0.f,0.f,0.f,0.f,0.f,0.f};
    #pragma unroll
    for (int i = 0; i < 16; ++i) {
        int d = (i << 8) + tid;                  // coalesced
        float xv = xr[d];
        ss += xv * xv;
        const float4 a0 = *(const float4*)(A + (size_t)d * 8);
        const float4 a1 = *(const float4*)(A + (size_t)d * 8 + 4);
        u[0] += xv * a0.x; u[1] += xv * a0.y; u[2] += xv * a0.z; u[3] += xv * a0.w;
        u[4] += xv * a1.x; u[5] += xv * a1.y; u[6] += xv * a1.z; u[7] += xv * a1.w;
    }
    #pragma unroll
    for (int r = 0; r < 8; ++r) red[tid][r] = u[r];
    red[tid][8] = ss;
    __syncthreads();
    if (tid < 9) {                               // deterministic column sums
        float s = 0.f;
        for (int i = 0; i < 256; ++i) s += red[i][tid];
        fin[tid] = s;
    }
    __syncthreads();
    if (tid == 0) {
        float sqrtK = kparams[1];
        float n  = sqrtf(fin[8]);
        float s1 = hyp_scale(n, sqrtK);
        float t_[8];
        #pragma unroll
        for (int r = 0; r < 8; ++r) t_[r] = s1 * fin[r];
        float nva2 = 0.f;
        #pragma unroll
        for (int r1 = 0; r1 < 8; ++r1)
            #pragma unroll
            for (int r2 = 0; r2 < 8; ++r2)
                nva2 += t_[r1] * G[r1 * 8 + r2] * t_[r2];
        float nva = sqrtf(fmaxf(nva2, 0.f));
        float s2  = hyp_scale(nva, sqrtK);
        #pragma unroll
        for (int r = 0; r < 8; ++r) tbuf[(size_t)row * 8 + r] = t_[r];
        srow[row] = 2.0f * s2;                   // SCALE = 16/8
    }
}

// ---------------- kernel 2: split-bf16 WMMA GEMM + fused bias/LoRA epilogue ----------------
struct SG { __bf16 Ahi[BM][KPAD]; __bf16 Alo[BM][KPAD]; __bf16 Bhi[BN][KPAD]; __bf16 Blo[BN][KPAD]; };
struct SE { float bias[BN]; float b2[8][BN]; float t[BM][8]; float s[BM]; };
union __align__(16) Smem { SG g; SE e; };

__device__ inline v16bf cat8(v8bf a, v8bf b) {
    return __builtin_shufflevector(a, b, 0,1,2,3,4,5,6,7,8,9,10,11,12,13,14,15);
}

__global__ __launch_bounds__(256, 1) void gemm_kernel(const float* __restrict__ x,
                                                      const float* __restrict__ W,
                                                      const float* __restrict__ bias,
                                                      const float* __restrict__ B2,
                                                      const float* __restrict__ tbuf,
                                                      const float* __restrict__ srow,
                                                      float* __restrict__ out) {
    __shared__ Smem sm;
    const int tid  = threadIdx.x;
    const int lane = tid & 31;
    const int wave = tid >> 5;
    const int wm   = wave >> 1;        // 0..3 -> 32-row band
    const int wn   = wave & 1;         // 0..1 -> 64-col band
    const int half = lane >> 4;
    const int l16  = lane & 15;
    const int m0 = blockIdx.y * BM;
    const int n0 = blockIdx.x * BN;

    v8f acc[2][4];
    #pragma unroll
    for (int mt = 0; mt < 2; ++mt)
        #pragma unroll
        for (int nt = 0; nt < 4; ++nt) acc[mt][nt] = (v8f)(0.0f);

    const int ldrow = tid >> 1;        // 0..127
    const int ldseg = tid & 1;         // 16-float segment

    const float* xbase = x + (size_t)(m0 + ldrow) * D_DIM + ldseg * 16;
    const float* wbase = W + (size_t)(n0 + ldrow) * D_DIM + ldseg * 16;

    // prologue: load K-tile 0 into registers
    float4 xq[4], wq[4];
    #pragma unroll
    for (int i = 0; i < 4; ++i) {
        xq[i] = *(const float4*)(xbase + 4 * i);
        wq[i] = *(const float4*)(wbase + 4 * i);
    }

    for (int kb = 0; kb < D_DIM; kb += BK) {
        // ---- convert current register tile to hi/lo bf16 and stage in LDS ----
        {
            const float* xf = (const float*)xq;
            const float* wf = (const float*)wq;
            __bf16 xh[16], xl[16], wh[16], wl[16];
            #pragma unroll
            for (int i = 0; i < 16; ++i) {
                float f = xf[i]; __bf16 h = (__bf16)f; xh[i] = h; xl[i] = (__bf16)(f - (float)h);
                float g = wf[i]; __bf16 k = (__bf16)g; wh[i] = k; wl[i] = (__bf16)(g - (float)k);
            }
            __bf16* ah = &sm.g.Ahi[ldrow][ldseg * 16];
            __bf16* al = &sm.g.Alo[ldrow][ldseg * 16];
            __bf16* bh = &sm.g.Bhi[ldrow][ldseg * 16];
            __bf16* bl = &sm.g.Blo[ldrow][ldseg * 16];
            #pragma unroll
            for (int i = 0; i < 2; ++i) {
                *(v8bf*)(ah + 8 * i) = *(v8bf*)(xh + 8 * i);
                *(v8bf*)(al + 8 * i) = *(v8bf*)(xl + 8 * i);
                *(v8bf*)(bh + 8 * i) = *(v8bf*)(wh + 8 * i);
                *(v8bf*)(bl + 8 * i) = *(v8bf*)(wl + 8 * i);
            }
        }
        __syncthreads();

        // ---- double-buffer: issue next tile's global loads now; they complete
        //      while the WMMAs below execute (latency hidden by compute) ----
        if (kb + BK < D_DIM) {
            const float* xp = xbase + kb + BK;
            const float* wp = wbase + kb + BK;
            #pragma unroll
            for (int i = 0; i < 4; ++i) {
                xq[i] = *(const float4*)(xp + 4 * i);
                wq[i] = *(const float4*)(wp + 4 * i);
            }
            if (kb + 2 * BK < D_DIM) {          // prime caches two tiles ahead
                __builtin_prefetch(xbase + kb + 2 * BK, 0, 1);
                __builtin_prefetch(wbase + kb + 2 * BK, 0, 1);
            }
        }

        // ---- fragment gathers (documented wave32 layouts) ----
        v16bf ahi[2], alo[2], bhi[4], blo[4];
        #pragma unroll
        for (int mt = 0; mt < 2; ++mt) {
            int ar = wm * 32 + mt * 16 + l16;
            const __bf16* ph = &sm.g.Ahi[ar][half * 8];   // K runs: [8h,8h+8) and [16+8h,...)
            const __bf16* pl = &sm.g.Alo[ar][half * 8];
            ahi[mt] = cat8(*(const v8bf*)ph, *(const v8bf*)(ph + 16));
            alo[mt] = cat8(*(const v8bf*)pl, *(const v8bf*)(pl + 16));
        }
        #pragma unroll
        for (int nt = 0; nt < 4; ++nt) {
            int br = wn * 64 + nt * 16 + l16;
            const __bf16* ph = &sm.g.Bhi[br][half * 8];   // K runs: [8h,8h+8) and [16+8h,...)
            const __bf16* pl = &sm.g.Blo[br][half * 8];
            bhi[nt] = cat8(*(const v8bf*)ph, *(const v8bf*)(ph + 16));
            blo[nt] = cat8(*(const v8bf*)pl, *(const v8bf*)(pl + 16));
        }

        // ---- split-bf16: hi*hi + hi*lo + lo*hi ≈ fp32 product ----
        #pragma unroll
        for (int mt = 0; mt < 2; ++mt)
            #pragma unroll
            for (int nt = 0; nt < 4; ++nt) {
                acc[mt][nt] = __builtin_amdgcn_wmma_f32_16x16x32_bf16(false, ahi[mt], false, bhi[nt], (short)0, acc[mt][nt], false, false);
                acc[mt][nt] = __builtin_amdgcn_wmma_f32_16x16x32_bf16(false, ahi[mt], false, blo[nt], (short)0, acc[mt][nt], false, false);
                acc[mt][nt] = __builtin_amdgcn_wmma_f32_16x16x32_bf16(false, alo[mt], false, bhi[nt], (short)0, acc[mt][nt], false, false);
            }
        __syncthreads();
    }

    // ---- epilogue staging: bias, LoRA-B tile, t rows, row scales ----
    if (tid < 128) {
        sm.e.bias[tid] = bias[n0 + tid];
        sm.e.s[tid]    = srow[m0 + tid];
    }
    #pragma unroll
    for (int i = tid; i < 1024; i += 256) {
        int r = i >> 7, c = i & 127;
        sm.e.b2[r][c] = B2[(size_t)r * D_DIM + n0 + c];
    }
    #pragma unroll
    for (int i = tid; i < 1024; i += 256) {
        int rr = i >> 3, r = i & 7;
        sm.e.t[rr][r] = tbuf[(size_t)(m0 + rr) * 8 + r];
    }
    __syncthreads();

    #pragma unroll
    for (int mt = 0; mt < 2; ++mt)
        #pragma unroll
        for (int nt = 0; nt < 4; ++nt) {
            int cl = wn * 64 + nt * 16 + l16;
            float bcol = sm.e.bias[cl];
            #pragma unroll
            for (int j = 0; j < 8; ++j) {
                int rl = wm * 32 + mt * 16 + j + 8 * half;
                float dot = 0.f;
                #pragma unroll
                for (int r = 0; r < 8; ++r) dot += sm.e.t[rl][r] * sm.e.b2[r][cl];
                float val = acc[mt][nt][j] + bcol + sm.e.s[rl] * dot;
                out[(size_t)(m0 + rl) * D_DIM + (n0 + cl)] = val;
            }
        }
}

extern "C" void kernel_launch(void* const* d_in, const int* in_sizes, int n_in,
                              void* d_out, int out_size, void* d_ws, size_t ws_size,
                              hipStream_t stream) {
    (void)in_sizes; (void)n_in; (void)out_size; (void)ws_size;
    const float* x    = (const float*)d_in[0];   // [8192, 4096]
    const float* W    = (const float*)d_in[1];   // [4096, 4096]
    const float* b    = (const float*)d_in[2];   // [4096]
    const float* A    = (const float*)d_in[3];   // [4096, 8]
    const float* B2   = (const float*)d_in[4];   // [8, 4096]
    const float* logK = (const float*)d_in[5];   // scalar
    float* out = (float*)d_out;

    float* kparams = (float*)d_ws;               // [2]
    float* G       = kparams + 16;               // [64]
    float* tbuf    = kparams + 128;              // [8192*8]
    float* srow    = tbuf + (size_t)M_DIM * 8;   // [8192]

    prep_kernel<<<1, 256, 0, stream>>>(B2, logK, G, kparams);
    row_kernel<<<M_DIM, 256, 0, stream>>>(x, A, G, kparams, tbuf, srow);
    dim3 grid(D_DIM / BN, M_DIM / BM);
    gemm_kernel<<<grid, 256, 0, stream>>>(x, W, b, B2, tbuf, srow, out);
}
